// UltraMemLayerV1_7670811590918
// MI455X (gfx1250) — matrix operand order repaired and checked by reading, not compiled, not run on hardware
//
#include <hip/hip_runtime.h>
#include <hip/hip_bf16.h>
#include <stdint.h>

typedef __attribute__((ext_vector_type(16))) __bf16 v16bf;
typedef __attribute__((ext_vector_type(8)))  float  v8f;
typedef __attribute__((address_space(3)))    char   lds_char;

#define BS        4096
#define HDIM      2048
#define KDIM_C    256
#define KEY_NUM   1024
#define HEADS     2
#define KNN_C     32
#define VALUE_NUM 262144
#define VDIM      256
#define EPS_C     1e-5f

union bf16x8  { uint4 q;    __bf16 e[8]; };
union bf16x16 { uint4 q[2]; v16bf v; __bf16 e[16]; };

// load 16 contiguous bf16 (2x16B) as a WMMA fragment vector
__device__ __forceinline__ v16bf load_frag16(const __bf16* p) {
  bf16x16 u;
  u.q[0] = *(const uint4*)p;
  u.q[1] = *(const uint4*)(p + 8);
  return u.v;
}
// build A fragment from two 8-element runs (K 16h+0..7 / 16h+8..15, contiguous)
__device__ __forceinline__ v16bf load_frag8x2(const __bf16* lo, const __bf16* hi) {
  bf16x16 u;
  u.q[0] = *(const uint4*)lo;
  u.q[1] = *(const uint4*)hi;
  return u.v;
}

// ---------------- CDNA5 async global->LDS copy (ASYNCcnt-tracked, §15.18.3)
__device__ __forceinline__ void async_copy_b64(const __bf16* gsrc, __bf16* ldst) {
  unsigned lds_addr = (unsigned)(uintptr_t)(lds_char*)(void*)ldst;
  asm volatile("global_load_async_to_lds_b64 %0, %1, off"
               :: "v"(lds_addr), "v"(gsrc)
               : "memory");
}
__device__ __forceinline__ void wait_async0() {
  asm volatile("s_wait_asynccnt 0x0" ::: "memory");
}

// ---------------------------------------------------------------- reductions
__device__ __forceinline__ float block_reduce_sum256(float v, float* scratch) {
  int lane = threadIdx.x & 31;
  int wv   = threadIdx.x >> 5;
#pragma unroll
  for (int off = 16; off > 0; off >>= 1) v += __shfl_xor(v, off, 32);
  if (lane == 0) scratch[wv] = v;
  __syncthreads();
  float tot = 0.f;
#pragma unroll
  for (int i = 0; i < 8; ++i) tot += scratch[i];
  __syncthreads();
  return tot;
}

// ------------------------------------------------ f32 -> bf16 (elementwise)
__global__ __launch_bounds__(256) void cvt_bf16_kernel(const float* __restrict__ src,
                                                       __bf16* __restrict__ dst) {
  const size_t i4 = ((size_t)blockIdx.x * 256 + threadIdx.x) * 4;
  float4 a = *(const float4*)&src[i4];
  __bf16 pk[4] = {(__bf16)a.x, (__bf16)a.y, (__bf16)a.z, (__bf16)a.w};
  *(uint2*)&dst[i4] = *(const uint2*)pk;
}

// --------------------------------------- f32 -> bf16 transposed panel (B^T)
__global__ __launch_bounds__(256) void transpose_bf16_kernel(const float* __restrict__ B,
                                                             __bf16* __restrict__ Bt,
                                                             int K, int N) {
  __shared__ float tile[32][33];
  const int k0 = blockIdx.x * 32;
  const int n0 = blockIdx.y * 32;
  const int tx = threadIdx.x & 31;
  const int ty = threadIdx.x >> 5;
#pragma unroll
  for (int j = 0; j < 4; ++j) {
    int r = ty + j * 8;
    tile[r][tx] = B[(size_t)(k0 + r) * N + (n0 + tx)];
  }
  __syncthreads();
#pragma unroll
  for (int j = 0; j < 4; ++j) {
    int r = ty + j * 8;  // n within tile
    Bt[(size_t)(n0 + r) * K + (k0 + tx)] = (__bf16)tile[tx][r];
  }
}

// ------------------------------------------------ WMMA GEMM (bf16 A, B^T)
// C(MxN) = A(MxK) @ B(KxN); A bf16 row-major, Bt[n][k] bf16. Block 32x128:
// 8 waves, each owns one 16-col slice and two 16-row tiles (2 WMMAs/k-step).
// A tile staged to LDS with CDNA5 async global->LDS copies.
__global__ __launch_bounds__(256) void gemm_bf16_kernel(const __bf16* __restrict__ A,
                                                        const __bf16* __restrict__ Bt,
                                                        float* __restrict__ C,
                                                        int M, int K, int N) {
  const int row0 = blockIdx.x * 32;
  const int col0 = blockIdx.y * 128;
  const int tid  = threadIdx.x;
  const int wv   = tid >> 5;
  const int lane = tid & 31;

  __shared__ __align__(16) __bf16 As[32][32];

  const int mrow  = lane & 15;
  const int koffA = (lane < 16) ? 0 : 16;
  const int ncol  = col0 + wv * 16 + (lane & 15);
  const __bf16* bp_base = Bt + (size_t)ncol * K + ((lane < 16) ? 0 : 16);

  // A staging coords: each thread async-copies 4 contiguous bf16 (8 bytes)
  const int sr = tid >> 3;        // 0..31 row
  const int sc = (tid & 7) * 4;   // 0..28 col

  v8f acc0 = {}, acc1 = {};
  for (int k0 = 0; k0 < K; k0 += 32) {
    async_copy_b64(&A[(size_t)(row0 + sr) * K + (k0 + sc)], &As[sr][sc]);
    wait_async0();
    __syncthreads();
    v16bf af0 = load_frag8x2(&As[mrow][koffA],      &As[mrow][8 + koffA]);
    v16bf af1 = load_frag8x2(&As[16 + mrow][koffA], &As[16 + mrow][8 + koffA]);
    v16bf bfr = load_frag16(bp_base + k0);
    acc0 = __builtin_amdgcn_wmma_f32_16x16x32_bf16(false, af0, false, bfr, (short)0, acc0,
                                                   false, false);
    acc1 = __builtin_amdgcn_wmma_f32_16x16x32_bf16(false, af1, false, bfr, (short)0, acc1,
                                                   false, false);
    __syncthreads();
  }

  const int rb = (lane < 16) ? 0 : 8;
#pragma unroll
  for (int j = 0; j < 8; ++j) {
    C[(size_t)(row0 + rb + j) * N + ncol]      = acc0[j];
    C[(size_t)(row0 + 16 + rb + j) * N + ncol] = acc1[j];
  }
}

// ------------------------------------------------------------ q layernorm
__global__ __launch_bounds__(256) void ln_q_kernel(const float* __restrict__ qraw,
                                                   const float* __restrict__ qw,
                                                   __bf16* __restrict__ qbf) {
  __shared__ float scratch[8];
  const int t = threadIdx.x;
  const size_t base = (size_t)blockIdx.x * 256;
  float x  = qraw[base + t];
  float s  = block_reduce_sum256(x, scratch);
  float ss = block_reduce_sum256(x * x, scratch);
  float m   = s * (1.f / 256.f);
  float var = ss * (1.f / 256.f) - m * m;
  float inv = rsqrtf(var + EPS_C);
  qbf[base + t] = (__bf16)((x - m) * inv * qw[t]);
}

// ------------------------------------------------------------ key norm
// keys (h,side,key,d,r) -> keysnb[h*2+side][key*2+r][d] bf16 (B^T layout)
__global__ __launch_bounds__(256) void keysnorm_kernel(const float* __restrict__ keys,
                                                       const float* __restrict__ kw,
                                                       __bf16* __restrict__ keysnb) {
  __shared__ float scratch[8];
  const int d  = threadIdx.x;
  const int k  = blockIdx.x;
  const int hs = blockIdx.y;
  const size_t base = ((size_t)hs * KEY_NUM + k) * KDIM_C * 2;
  float x0 = keys[base + (size_t)d * 2 + 0];
  float x1 = keys[base + (size_t)d * 2 + 1];
  float s0 = block_reduce_sum256(x0, scratch);
  float q0 = block_reduce_sum256(x0 * x0, scratch);
  float s1 = block_reduce_sum256(x1, scratch);
  float q1 = block_reduce_sum256(x1 * x1, scratch);
  float m0 = s0 * (1.f / 256.f), v0 = q0 * (1.f / 256.f) - m0 * m0;
  float m1 = s1 * (1.f / 256.f), v1 = q1 * (1.f / 256.f) - m1 * m1;
  float w = kw[d];
  const size_t ob = (size_t)hs * (KEY_NUM * 2) * KDIM_C;
  keysnb[ob + (size_t)(k * 2 + 0) * KDIM_C + d] = (__bf16)((x0 - m0) * rsqrtf(v0 + EPS_C) * w);
  keysnb[ob + (size_t)(k * 2 + 1) * KDIM_C + d] = (__bf16)((x1 - m1) * rsqrtf(v1 + EPS_C) * w);
}

// ----------------------------------------- scores + tucker reduce + top-32
__global__ __launch_bounds__(256) void scores_topk_kernel(const __bf16* __restrict__ qbf,
                                                          const __bf16* __restrict__ keysnb,
                                                          const float* __restrict__ tucker_u,
                                                          const float* __restrict__ tucker_v,
                                                          int* __restrict__ i1, int* __restrict__ i2,
                                                          float* __restrict__ g1, float* __restrict__ g2) {
  const int bt   = blockIdx.x;
  const int h    = blockIdx.y;
  const int side = blockIdx.z;
  const int tid  = threadIdx.x;
  const int wv   = tid >> 5;
  const int lane = tid & 31;

  __shared__ float sred[16][KEY_NUM];  // 64 KB (CDNA5 WGP: 320 KB LDS)
  __shared__ int   sel[16][KNN_C];

  const float* uvp = (side == 0) ? tucker_u : tucker_v;
  const float u0 = uvp[h * 2 + 0];
  const float u1 = uvp[h * 2 + 1];
  int*   iout = (side == 0) ? i1 : i2;
  float* gout = (side == 0) ? g1 : g2;

  const __bf16* qb = qbf + ((size_t)(bt * 16) * 2 + side) * KDIM_C;  // row stride 512
  const __bf16* kb = keysnb + (size_t)(h * 2 + side) * (KEY_NUM * 2) * KDIM_C;

  // A fragments for all 8 K-steps (invariant across column tiles)
  v16bf afs[8];
  {
    const int mrow  = lane & 15;
    const int koffA = (lane < 16) ? 0 : 16;
    const __bf16* ap0 = qb + (size_t)mrow * (2 * KDIM_C);
#pragma unroll
    for (int ks = 0; ks < 8; ++ks) {
      const __bf16* ap = ap0 + ks * 32 + koffA;
      afs[ks] = load_frag8x2(ap, ap + 8);
    }
  }

  for (int nt = wv; nt < 128; nt += 8) {
    const int n0   = nt * 16;
    const int ncol = n0 + (lane & 15);
    v8f acc = {};
    const __bf16* bp0 = kb + (size_t)ncol * KDIM_C + ((lane < 16) ? 0 : 16);
#pragma unroll
    for (int ks = 0; ks < 8; ++ks) {
      v16bf bfr = load_frag16(bp0 + ks * 32);
      acc = __builtin_amdgcn_wmma_f32_16x16x32_bf16(false, afs[ks], false, bfr, (short)0, acc,
                                                    false, false);
    }
    // tucker rank reduce: adjacent columns are (r0,r1) of the same key
#pragma unroll
    for (int j = 0; j < 8; ++j) {
      float other = __shfl_xor(acc[j], 1, 32);
      if ((lane & 1) == 0) {
        sred[((lane < 16) ? 0 : 8) + j][ncol >> 1] = acc[j] * u0 + other * u1;
      }
    }
  }
  __syncthreads();

  // per-row top-32 over 1024 keys; wave wv handles rows 2*wv, 2*wv+1
  for (int rr = 0; rr < 2; ++rr) {
    const int row = wv * 2 + rr;
    unsigned mask = 0;
    for (int pass = 0; pass < KNN_C; ++pass) {
      float best = -3.0e38f;
      int   bi   = -1;
#pragma unroll 4
      for (int i = 0; i < 32; ++i) {
        if (mask & (1u << i)) continue;
        float v = sred[row][i * 32 + lane];
        if (v > best) { best = v; bi = i; }
      }
      int bk = (bi < 0) ? lane : (bi * 32 + lane);
#pragma unroll
      for (int off = 16; off > 0; off >>= 1) {
        float ov = __shfl_xor(best, off, 32);
        int   ok = __shfl_xor(bk, off, 32);
        if (ov > best) { best = ov; bk = ok; }
      }
      if ((bk & 31) == lane) mask |= 1u << (bk >> 5);
      if (lane == 0) sel[row][pass] = bk;
    }
  }
  __syncthreads();

  // gather rank-resolved g for selected keys (vectorized 8-wide bf16 dots)
  for (int t = tid; t < 16 * KNN_C * 2; t += 256) {
    const int r   = t & 1;
    const int j   = (t >> 1) & 31;
    const int row = t >> 6;
    const int key = sel[row][j];
    const __bf16* kp = kb + (size_t)(key * 2 + r) * KDIM_C;
    const __bf16* qp = qb + (size_t)row * (2 * KDIM_C);
    float s = 0.f;
    for (int d = 0; d < KDIM_C; d += 8) {
      bf16x8 qa, ka;
      qa.q = *(const uint4*)(qp + d);
      ka.q = *(const uint4*)(kp + d);
#pragma unroll
      for (int e = 0; e < 8; ++e) s += (float)qa.e[e] * (float)ka.e[e];
    }
    const int b = bt * 16 + row;
    gout[(((size_t)b * HEADS + h) * KNN_C + j) * 2 + r] = s;
    if (r == 0) iout[((size_t)b * HEADS + h) * KNN_C + j] = key;
  }
}

// ------------------------------- combine: bilinear scores, top-32, softmax,
//                                  value gather + expert-slot accumulation
__global__ __launch_bounds__(256) void combine_kernel(const int* __restrict__ i1,
                                                      const int* __restrict__ i2,
                                                      const float* __restrict__ g1,
                                                      const float* __restrict__ g2,
                                                      const float* __restrict__ core,
                                                      const int* __restrict__ shuffle_index,
                                                      const float* __restrict__ values,
                                                      __bf16* __restrict__ aggb) {
  const int b    = blockIdx.x;
  const int tid  = threadIdx.x;
  const int wv   = tid >> 5;
  const int lane = tid & 31;

  __shared__ float sc[1024];
  __shared__ int   sidx[1024];
  __shared__ float aggsh[4][VDIM];
  __shared__ float wsel[KNN_C];
  __shared__ int   isel[KNN_C];

  for (int t = tid; t < 4 * VDIM; t += 256) aggsh[t >> 8][t & 255] = 0.f;

  for (int h = 0; h < HEADS; ++h) {
    const float c00 = core[h * 4 + 0] + core[8 + h * 4 + 0];
    const float c01 = core[h * 4 + 1] + core[8 + h * 4 + 1];
    const float c10 = core[h * 4 + 2] + core[8 + h * 4 + 2];
    const float c11 = core[h * 4 + 3] + core[8 + h * 4 + 3];
    const float* G1 = g1 + ((size_t)b * HEADS + h) * (KNN_C * 2);
    const float* G2 = g2 + ((size_t)b * HEADS + h) * (KNN_C * 2);
    const int*   I1 = i1 + ((size_t)b * HEADS + h) * KNN_C;
    const int*   I2 = i2 + ((size_t)b * HEADS + h) * KNN_C;
    __syncthreads();
    for (int t = tid; t < 1024; t += 256) {
      const int k = t >> 5, l = t & 31;
      const float a0 = G1[k * 2 + 0], a1 = G1[k * 2 + 1];
      const float b0 = G2[l * 2 + 0], b1 = G2[l * 2 + 1];
      sc[t]   = a0 * (c00 * b0 + c01 * b1) + a1 * (c10 * b0 + c11 * b1);
      sidx[t] = I1[k] * KEY_NUM + I2[l];
    }
    __syncthreads();

    if (wv == 0) {
      unsigned mask = 0;
      for (int pass = 0; pass < KNN_C; ++pass) {
        float best = -3.0e38f;
        int   bi   = -1;
#pragma unroll 4
        for (int i = 0; i < 32; ++i) {
          if (mask & (1u << i)) continue;
          float v = sc[i * 32 + lane];
          if (v > best) { best = v; bi = i; }
        }
        int bt = (bi < 0) ? lane : (bi * 32 + lane);
#pragma unroll
        for (int off = 16; off > 0; off >>= 1) {
          float ov = __shfl_xor(best, off, 32);
          int   ot = __shfl_xor(bt, off, 32);
          if (ov > best) { best = ov; bt = ot; }
        }
        if ((bt & 31) == lane) mask |= 1u << (bt >> 5);
        if (lane == 0) { wsel[pass] = best; isel[pass] = sidx[bt]; }
      }
      float myv = wsel[lane];
      float mx  = __shfl(myv, 0, 32);
      float e   = __expf(myv - mx);
      float ssum = e;
#pragma unroll
      for (int off = 16; off > 0; off >>= 1) ssum += __shfl_xor(ssum, off, 32);
      wsel[lane] = e / ssum;
    }
    __syncthreads();

    for (int j = 0; j < KNN_C; ++j) {
      const int shuf = shuffle_index[isel[j]];
      const int vidx = shuf & (VALUE_NUM - 1);
      const int eidx = shuf >> 18;
      if (j + 1 < KNN_C) {
        const int shuf2 = shuffle_index[isel[j + 1]];
        __builtin_prefetch(values + (size_t)(shuf2 & (VALUE_NUM - 1)) * VDIM, 0, 1);
      }
      aggsh[eidx][tid] += wsel[j] * values[(size_t)vidx * VDIM + tid];
    }
  }
  __syncthreads();
  for (int t = tid; t < 4 * VDIM; t += 256)
    aggb[(size_t)b * 1024 + t] = (__bf16)aggsh[t >> 8][t & 255];
}

// -------------------------------------------------------------------- launch
extern "C" void kernel_launch(void* const* d_in, const int* in_sizes, int n_in,
                              void* d_out, int out_size, void* d_ws, size_t ws_size,
                              hipStream_t stream) {
  const float* hidden  = (const float*)d_in[0];
  const float* wq      = (const float*)d_in[1];
  const float* qnw     = (const float*)d_in[2];
  const float* knw     = (const float*)d_in[3];
  const float* keys    = (const float*)d_in[4];
  const float* core    = (const float*)d_in[5];
  const float* tu      = (const float*)d_in[6];
  const float* tv      = (const float*)d_in[7];
  const float* values  = (const float*)d_in[8];
  const float* wv      = (const float*)d_in[9];
  const int*   shuffle = (const int*)d_in[10];
  float* out = (float*)d_out;

  char* p = (char*)d_ws;
  auto take = [&](size_t bytes) {
    char* r = p;
    p += (bytes + 255) & ~(size_t)255;
    return r;
  };
  __bf16* hiddenb = (__bf16*)take((size_t)BS * HDIM * 2);
  float*  qraw    = (float*)take((size_t)BS * 512 * 4);
  __bf16* qbf     = (__bf16*)take((size_t)BS * 512 * 2);
  __bf16* keysnb  = (__bf16*)take((size_t)4 * KEY_NUM * 2 * KDIM_C * 2);
  __bf16* wqT     = (__bf16*)take((size_t)512 * HDIM * 2);
  __bf16* wvT     = (__bf16*)take((size_t)HDIM * 1024 * 2);
  int*    i1      = (int*)take((size_t)BS * HEADS * KNN_C * 4);
  int*    i2      = (int*)take((size_t)BS * HEADS * KNN_C * 4);
  float*  g1      = (float*)take((size_t)BS * HEADS * KNN_C * 2 * 4);
  float*  g2      = (float*)take((size_t)BS * HEADS * KNN_C * 2 * 4);
  __bf16* aggb    = (__bf16*)take((size_t)BS * 1024 * 2);

  // 0) bf16 operand prep (hidden, transposed weight panels)
  cvt_bf16_kernel<<<(BS * HDIM) / 1024, 256, 0, stream>>>(hidden, hiddenb);
  transpose_bf16_kernel<<<dim3(HDIM / 32, 512 / 32), 256, 0, stream>>>(wq, wqT, HDIM, 512);
  transpose_bf16_kernel<<<dim3(1024 / 32, HDIM / 32), 256, 0, stream>>>(wv, wvT, 1024, HDIM);
  // 1) q = hidden @ wq  (4096 x 2048 x 512)
  gemm_bf16_kernel<<<dim3(BS / 32, 512 / 128), 256, 0, stream>>>(hiddenb, wqT, qraw, BS, HDIM, 512);
  // 2) layernorm -> bf16 q
  ln_q_kernel<<<BS * 2, 256, 0, stream>>>(qraw, qnw, qbf);
  // 3) key normalization -> bf16 B^T layout
  keysnorm_kernel<<<dim3(KEY_NUM, 4), 256, 0, stream>>>(keys, knw, keysnb);
  // 4) WMMA scores + tucker reduce + per-row top-32 + g gather
  scores_topk_kernel<<<dim3(BS / 16, HEADS, 2), 256, 0, stream>>>(qbf, keysnb, tu, tv, i1, i2,
                                                                  g1, g2);
  // 5) bilinear combine + top-32 + softmax + value gather -> bf16 agg
  combine_kernel<<<BS, 256, 0, stream>>>(i1, i2, g1, g2, core, shuffle, values, aggb);
  // 6) out = agg @ wv  (4096 x 1024 x 2048)
  gemm_bf16_kernel<<<dim3(BS / 32, HDIM / 128), 256, 0, stream>>>(aggb, wvT, out, BS, 1024, HDIM);
}
